// SpectralConv1d_15006615734080
// MI455X (gfx1250) — compile-verified, matrix operand order checked by reading, not run on hardware
//
#include <hip/hip_runtime.h>
#include <hip/hip_bf16.h>

typedef __attribute__((ext_vector_type(16))) __bf16 v16bf;
typedef __attribute__((ext_vector_type(8)))  float  v8f;
typedef __attribute__((ext_vector_type(4)))  unsigned int u32x4;
typedef __attribute__((ext_vector_type(8)))  int i32x8;
typedef __attribute__((ext_vector_type(4)))  int i32x4;

#define BSZ    32
#define CIN    128
#define COUT   128
#define MODES  32
#define NPTS   8192
#define MP     64              // 2*MODES (re/im interleaved columns)
#define ROWS   4096            // BSZ*CIN == BSZ*COUT
#define KSPLIT 16
#define KCHUNK (NPTS / KSPLIT) // 512
#define LDSPITCH 520           // 512 + 8 ushorts (16B pad per row -> conflict-free b128 reads)

__device__ __forceinline__ unsigned short f2bf(float f) {
    unsigned int u = __float_as_uint(f);
    u += 0x7FFFu + ((u >> 16) & 1u);      // round-to-nearest-even
    return (unsigned short)(u >> 16);
}

union FragBF {
    uint4          q[2];
    unsigned short s[16];
    v16bf          v;
};

// ---------------------------------------------------------------------------
// Basis tables: col c (even -> cos(2*pi*m*n/N), odd -> -sin), m = c/2.
// b1 : [c][n]  (K=n contiguous)  for GEMM1 B-fragments
// b3 : [n][c]  (K=c contiguous)  for GEMM3 B-fragments
// ---------------------------------------------------------------------------
__global__ void basis_kernel(unsigned short* __restrict__ b1,
                             unsigned short* __restrict__ b3) {
    int idx = blockIdx.x * blockDim.x + threadIdx.x;   // 64*8192 threads
    int c = idx >> 13;
    int n = idx & (NPTS - 1);
    int m = c >> 1;
    int t = (m * n) & (NPTS - 1);
    float th = (float)t * (6.28318530717958647692f / (float)NPTS);
    float sv, cv;
    __sincosf(th, &sv, &cv);
    float val = (c & 1) ? -sv : cv;
    unsigned short bv = f2bf(val);
    b1[(size_t)c * NPTS + n] = bv;
    b3[(size_t)n * MP + c]   = bv;
}

// w[i][o][m] fp32 -> wbf[m][o][i] bf16  (K=i contiguous for GEMM2 B-frags)
__global__ void wpack_kernel(const float* __restrict__ w,
                             unsigned short* __restrict__ wbf) {
    int idx = blockIdx.x * blockDim.x + threadIdx.x;   // 32*128*128 threads
    int m = idx >> 14;
    int o = (idx >> 7) & (COUT - 1);
    int i = idx & (CIN - 1);
    wbf[idx] = f2bf(w[(size_t)i * (COUT * MODES) + (size_t)o * MODES + m]);
}

// ---------------------------------------------------------------------------
// GEMM1: Xpart[s][row][c] = sum_{n in chunk s} x[row][n] * F[n][c]
// Block = 8 waves sharing one k-split s; basis chunk [64][512] bf16 staged
// into LDS (row pitch 520) by the Tensor Data Mover, consumed via ds_load.
// A (x, fp32) loads are software-pipelined one k-step ahead of the WMMAs.
// ---------------------------------------------------------------------------
__global__ void __launch_bounds__(256)
gemm1_kernel(const float* __restrict__ x,
             const unsigned short* __restrict__ b1,
             float* __restrict__ xpart) {
    __shared__ unsigned short ldsb[64 * LDSPITCH];     // 66560 bytes

    int tid  = threadIdx.x;
    int widx = tid >> 5;
    int lane = tid & 31;
    int r = lane & 15, h = lane >> 4;

    int s       = blockIdx.x & (KSPLIT - 1);
    int rowtile = ((blockIdx.x >> 4) << 3) + widx;     // 0..255
    int rowbase = rowtile << 4;
    int kbase   = s * KCHUNK;

    // ---- TDM: stage b1[0..63][kbase .. kbase+512) into padded LDS --------
    if (tid == 0) {
        unsigned lds_addr = (unsigned)(unsigned long long)(&ldsb[0]);
        unsigned long long ga = (unsigned long long)(const void*)(b1 + kbase);
        u32x4 g0;
        g0[0] = 1u;                                       // count=1 descriptor
        g0[1] = lds_addr;                                 // lds_addr[31:0]
        g0[2] = (unsigned)(ga & 0xFFFFFFFFull);           // global_addr[31:0]
        g0[3] = (unsigned)((ga >> 32) & 0x01FFFFFFull)    // global_addr[56:32]
              | 0x80000000u;                              // type=2 (image)
        i32x8 g1;
        // data_size=1(2B) | pad_enable | pad_interval=7(256 DW) | pad_amount=3(4 DW)
        g1[0] = (int)0x07D10000u;
        g1[1] = (int)((NPTS & 0xFFFFu) << 16);            // tensor_dim0 lo16
        g1[2] = (int)(((NPTS >> 16) & 0xFFFFu) | (64u << 16)); // dim0 hi / tensor_dim1 lo
        g1[3] = (int)(0u | ((unsigned)KCHUNK << 16));     // tensor_dim1 hi / tile_dim0=512
        g1[4] = 64;                                       // tile_dim1=64, tile_dim2=0
        g1[5] = NPTS;                                     // tensor_dim0_stride lo32
        g1[6] = 0;                                        // stride hi / dim1_stride lo
        g1[7] = 0;
        i32x4 g2 = {0, 0, 0, 0};
        i32x4 g3 = {0, 0, 0, 0};
        i32x8 g4 = {0, 0, 0, 0, 0, 0, 0, 0};
        __builtin_amdgcn_tensor_load_to_lds(g0, g1, g2, g3, g4, 0);
        __builtin_amdgcn_s_wait_tensorcnt(0);
    }
    __syncthreads();

    v8f acc[4] = {};
    const float* xrow = x + (size_t)(rowbase + r) * NPTS + kbase;

    // software pipeline: preload A floats for k-step 0
    float4 f0 = *(const float4*)(xrow + (h << 3));
    float4 f1 = *(const float4*)(xrow + (h << 3) + 4);
    float4 f2 = *(const float4*)(xrow + (h << 3) + 16);
    float4 f3 = *(const float4*)(xrow + (h << 3) + 20);

    for (int kt = 0; kt < KCHUNK / 32; ++kt) {
        FragBF a;
        a.s[0]=f2bf(f0.x); a.s[1]=f2bf(f0.y); a.s[2]=f2bf(f0.z); a.s[3]=f2bf(f0.w);
        a.s[4]=f2bf(f1.x); a.s[5]=f2bf(f1.y); a.s[6]=f2bf(f1.z); a.s[7]=f2bf(f1.w);
        a.s[8]=f2bf(f2.x); a.s[9]=f2bf(f2.y); a.s[10]=f2bf(f2.z); a.s[11]=f2bf(f2.w);
        a.s[12]=f2bf(f3.x); a.s[13]=f2bf(f3.y); a.s[14]=f2bf(f3.z); a.s[15]=f2bf(f3.w);

        if (kt + 1 < KCHUNK / 32) {                       // prefetch next A
            int l0 = ((kt + 1) << 5) + (h << 3);
            f0 = *(const float4*)(xrow + l0);
            f1 = *(const float4*)(xrow + l0 + 4);
            f2 = *(const float4*)(xrow + l0 + 16);
            f3 = *(const float4*)(xrow + l0 + 20);
        }

        int l0 = (kt << 5) + (h << 3);
        #pragma unroll
        for (int ct = 0; ct < 4; ++ct) {
            const unsigned short* brow = &ldsb[(size_t)((ct << 4) + r) * LDSPITCH];
            FragBF bf;
            bf.q[0] = *(const uint4*)(brow + l0);
            bf.q[1] = *(const uint4*)(brow + l0 + 16);
            acc[ct] = __builtin_amdgcn_wmma_f32_16x16x32_bf16(
                false, a.v, false, bf.v, (short)0, acc[ct], false, false);
        }
    }

    float* base = xpart + (size_t)s * (ROWS * MP);
    #pragma unroll
    for (int ct = 0; ct < 4; ++ct)
        #pragma unroll
        for (int v = 0; v < 8; ++v)
            base[(size_t)(rowbase + v + (h << 3)) * MP + (ct << 4) + r] = acc[ct][v];
}

// Reduce split-K partials, pack bf16 into xbf[c][row]  (row = b*128 + i)
__global__ void reduce_kernel(const float* __restrict__ xpart,
                              unsigned short* __restrict__ xbf) {
    int idx = blockIdx.x * blockDim.x + threadIdx.x;   // 4096*64 threads
    int row = idx >> 6;
    int c   = idx & (MP - 1);
    float s = 0.f;
    #pragma unroll
    for (int p = 0; p < KSPLIT; ++p)
        s += xpart[(size_t)p * (ROWS * MP) + idx];
    xbf[(size_t)c * ROWS + row] = f2bf(s);
}

// ---------------------------------------------------------------------------
// GEMM2: for each mode-part mp: Y[b][o] = sum_i X[mp][b][i] * W[m][o][i]
// irfft scale (m==0?1:2)/N folded into bf16 output ybf[b][o][mp]
// ---------------------------------------------------------------------------
__global__ void __launch_bounds__(256)
gemm2_kernel(const unsigned short* __restrict__ xbf,
             const unsigned short* __restrict__ wbf,
             unsigned short* __restrict__ ybf) {
    int tid  = threadIdx.x;
    int wave = (blockIdx.x << 3) + (tid >> 5);         // 1024 waves
    int lane = tid & 31;
    int r = lane & 15, h = lane >> 4;
    int mp = wave >> 4;
    int bt = (wave >> 3) & 1;
    int ot = wave & 7;
    int m  = mp >> 1;

    const unsigned short* arow = xbf + (size_t)mp * ROWS + (size_t)((bt << 4) + r) * CIN;
    const unsigned short* brow = wbf + (size_t)m * (COUT * CIN) + (size_t)((ot << 4) + r) * CIN;

    v8f acc = {};
    #pragma unroll
    for (int kt = 0; kt < CIN / 32; ++kt) {
        int o1 = (kt << 5) + (h << 3);
        int o2 = o1 + 16;
        FragBF a, b;
        a.q[0] = *(const uint4*)(arow + o1); a.q[1] = *(const uint4*)(arow + o2);
        b.q[0] = *(const uint4*)(brow + o1); b.q[1] = *(const uint4*)(brow + o2);
        acc = __builtin_amdgcn_wmma_f32_16x16x32_bf16(
            false, a.v, false, b.v, (short)0, acc, false, false);
    }

    float scale = (m == 0 ? 1.0f : 2.0f) / (float)NPTS;
    #pragma unroll
    for (int v = 0; v < 8; ++v) {
        int brow_ = (bt << 4) + v + (h << 3);
        int o     = (ot << 4) + r;
        ybf[(size_t)(brow_ * COUT + o) * MP + mp] = f2bf(scale * acc[v]);
    }
}

// ---------------------------------------------------------------------------
// GEMM3: out[row][n] = sum_c Y'[row][c] * F[c][n] + bias[row%COUT]
// one wave = 16 rows x 128 cols (8 col-tiles), K = 64; B double-buffered
// ---------------------------------------------------------------------------
__global__ void __launch_bounds__(256)
gemm3_kernel(const unsigned short* __restrict__ ybf,
             const unsigned short* __restrict__ b3,
             const float* __restrict__ bias,
             float* __restrict__ out) {
    int tid  = threadIdx.x;
    int wave = (blockIdx.x << 3) + (tid >> 5);         // 16384 waves
    int lane = tid & 31;
    int r = lane & 15, h = lane >> 4;
    int rowtile = wave >> 6;
    int cg      = wave & 63;
    int rowbase = rowtile << 4;
    int colbase = cg << 7;

    const unsigned short* arow = ybf + (size_t)(rowbase + r) * MP;
    FragBF a0, a1;
    a0.q[0] = *(const uint4*)(arow + (h << 3));
    a0.q[1] = *(const uint4*)(arow + 16 + (h << 3));
    a1.q[0] = *(const uint4*)(arow + 32 + (h << 3));
    a1.q[1] = *(const uint4*)(arow + 48 + (h << 3));

    FragBF bb[2][2];
    auto loadB = [&](int ct, FragBF* dst) {
        const unsigned short* brow = b3 + (size_t)(colbase + (ct << 4) + r) * MP;
        dst[0].q[0] = *(const uint4*)(brow + (h << 3));
        dst[0].q[1] = *(const uint4*)(brow + 16 + (h << 3));
        dst[1].q[0] = *(const uint4*)(brow + 32 + (h << 3));
        dst[1].q[1] = *(const uint4*)(brow + 48 + (h << 3));
    };
    loadB(0, bb[0]);

    #pragma unroll
    for (int ct = 0; ct < 8; ++ct) {
        if (ct + 1 < 8) loadB(ct + 1, bb[(ct + 1) & 1]);

        v8f acc = {};
        acc = __builtin_amdgcn_wmma_f32_16x16x32_bf16(
            false, a0.v, false, bb[ct & 1][0].v, (short)0, acc, false, false);
        acc = __builtin_amdgcn_wmma_f32_16x16x32_bf16(
            false, a1.v, false, bb[ct & 1][1].v, (short)0, acc, false, false);

        #pragma unroll
        for (int v = 0; v < 8; ++v) {
            int row = rowbase + v + (h << 3);
            out[(size_t)row * NPTS + colbase + (ct << 4) + r] =
                acc[v] + bias[row & (COUT - 1)];
        }
    }
}

// ---------------------------------------------------------------------------
extern "C" void kernel_launch(void* const* d_in, const int* in_sizes, int n_in,
                              void* d_out, int out_size, void* d_ws, size_t ws_size,
                              hipStream_t stream) {
    const float* x    = (const float*)d_in[0];   // [32][128][8192]
    const float* w    = (const float*)d_in[1];   // [128][128][32]
    const float* bias = (const float*)d_in[2];   // [128]
    float*       out  = (float*)d_out;           // [32][128][8192]

    char* ws = (char*)d_ws;
    const size_t OFF_XPART = 0;                                           // 16 MB fp32
    const size_t OFF_XBF   = OFF_XPART + (size_t)KSPLIT * ROWS * MP * 4;  // 512 KB
    const size_t OFF_WBF   = OFF_XBF + (size_t)MP * ROWS * 2;             // 1 MB
    const size_t OFF_YBF   = OFF_WBF + (size_t)MODES * COUT * CIN * 2;    // 512 KB
    const size_t OFF_B1    = OFF_YBF + (size_t)ROWS * MP * 2;             // 1 MB
    const size_t OFF_B3    = OFF_B1 + (size_t)MP * NPTS * 2;              // 1 MB
    const size_t TOTAL     = OFF_B3 + (size_t)NPTS * MP * 2;

    if (ws_size < TOTAL) return;  // scratch too small: cannot run safely

    float*          xpart = (float*)(ws + OFF_XPART);
    unsigned short* xbf   = (unsigned short*)(ws + OFF_XBF);
    unsigned short* wbf   = (unsigned short*)(ws + OFF_WBF);
    unsigned short* ybf   = (unsigned short*)(ws + OFF_YBF);
    unsigned short* b1    = (unsigned short*)(ws + OFF_B1);
    unsigned short* b3    = (unsigned short*)(ws + OFF_B3);

    basis_kernel <<<(MP * NPTS) / 256, 256, 0, stream>>>(b1, b3);
    wpack_kernel <<<(MODES * COUT * CIN) / 256, 256, 0, stream>>>(w, wbf);
    gemm1_kernel <<<(256 * KSPLIT) / 8, 256, 0, stream>>>(x, b1, xpart);
    reduce_kernel<<<(ROWS * MP) / 256, 256, 0, stream>>>(xpart, xbf);
    gemm2_kernel <<<1024 / 8, 256, 0, stream>>>(xbf, wbf, ybf);
    gemm3_kernel <<<16384 / 8, 256, 0, stream>>>(ybf, b3, bias, out);
}
